// ScaledDotProductAttention_57277683860022
// MI455X (gfx1250) — matrix operand order repaired and checked by reading, not compiled, not run on hardware
//
#include <hip/hip_runtime.h>
#include <hip/hip_bf16.h>
#include <stdint.h>

// ---------------------------------------------------------------------------
// Flash-attention forward, causal, fp32 in/out, bf16 WMMA compute (gfx1250).
// B=2 H=16 S=2048 D=128.  64-key softmax chunks, pair-packed transposed V.
// ---------------------------------------------------------------------------

typedef __attribute__((ext_vector_type(16))) __bf16 v16bf;
typedef __attribute__((ext_vector_type(8)))  float  v8f;

#define S_LEN   2048
#define D_DIM   128
#define QTILE   128
#define KTILE   128
#define PADW    136                  // 128 + 8 bf16 pad -> 272B row stride (16B aligned)
#define NEGBIG  (-1.0e9f)
#define LOG2E   1.4426950408889634f
#define SM_SCALE 0.08838834764831845f   // 1/sqrt(128)

__device__ __forceinline__ unsigned f2bf1(float x) {
  // round-to-nearest-even f32 -> bf16 (bit trick)
  unsigned u = __builtin_bit_cast(unsigned, x);
  return (u + 0x7FFFu + ((u >> 16) & 1u)) >> 16;
}
__device__ __forceinline__ unsigned pkbf(float a, float b) {
  return f2bf1(a) | (f2bf1(b) << 16);
}

union FragBF { v16bf v; unsigned u[8]; uint4 q[2]; };

// A fragment, this code's contiguous k order: lanes h=0 hold K = d0+0..15,
//                           lanes h=1 hold K = d0+16..31
__device__ __forceinline__ v16bf ldsFragA(const unsigned short* s, int row, int d0, int h) {
  FragBF f;
  const unsigned short* p = s + row * PADW;
  f.q[0] = *(const uint4*)(p + d0 + 16 * h);
  f.q[1] = *(const uint4*)(p + d0 + 16 * h + 8);
  return f.v;
}
// B-matrix 16-bit striping: lane column n = lane&15, K = {8h..8h+7, 16+8h..16+8h+7}
__device__ __forceinline__ v16bf ldsFragB(const unsigned short* s, int row, int e0) {
  FragBF f;
  const unsigned short* p = s + row * PADW + e0;
  f.q[0] = *(const uint4*)(p);
  f.q[1] = *(const uint4*)(p + 16);
  return f.v;
}

__global__ __launch_bounds__(256)
void fa_fwd_causal(const float* __restrict__ Qg, const float* __restrict__ Kg,
                   const float* __restrict__ Vg, float* __restrict__ Og)
{
  __shared__ __align__(16) unsigned short sK [KTILE * PADW];   // [key][d]  bf16
  __shared__ __align__(16) unsigned short sVt[D_DIM * PADW];   // [d][key]  bf16 (transposed)

  const int qtile = blockIdx.x;
  const int bh    = blockIdx.y;
  const int tid   = threadIdx.x;
  const int lane  = tid & 31;
  const int w     = tid >> 5;          // wave id 0..7
  const int h     = lane >> 4;         // half-wave
  const int lr    = lane & 15;
  const int sb    = 8 * h;

  const size_t base = (size_t)bh * S_LEN * D_DIM;
  const int q_local = w * 16 + lr;                 // 0..127 within tile
  const int q_abs   = qtile * QTILE + q_local;

  // ---- Q fragments (B-pattern: 16 contiguous d-elems at d0 + 16h), kept in
  //      VGPRs; softmax scale 1/sqrt(d) folded into Q here.
  FragBF qf[4];
  {
    const float* qr = Qg + base + (size_t)q_abs * D_DIM;
#pragma unroll
    for (int j = 0; j < 4; ++j) {
      const float4* p = (const float4*)(qr + 32 * j + 16 * h);
      float4 f0 = p[0], f1 = p[1], f2 = p[2], f3 = p[3];
      qf[j].u[0] = pkbf(f0.x * SM_SCALE, f0.y * SM_SCALE);
      qf[j].u[1] = pkbf(f0.z * SM_SCALE, f0.w * SM_SCALE);
      qf[j].u[2] = pkbf(f1.x * SM_SCALE, f1.y * SM_SCALE);
      qf[j].u[3] = pkbf(f1.z * SM_SCALE, f1.w * SM_SCALE);
      qf[j].u[4] = pkbf(f2.x * SM_SCALE, f2.y * SM_SCALE);
      qf[j].u[5] = pkbf(f2.z * SM_SCALE, f2.w * SM_SCALE);
      qf[j].u[6] = pkbf(f3.x * SM_SCALE, f3.y * SM_SCALE);
      qf[j].u[7] = pkbf(f3.z * SM_SCALE, f3.w * SM_SCALE);
    }
  }

  v8f zero = {};
  v8f o[8];
#pragma unroll
  for (int i = 0; i < 8; ++i) o[i] = zero;
  float m = -1.0e30f, lsum = 0.0f;

  for (int kb = 0; kb <= qtile; ++kb) {
    __syncthreads();
    // ---- stage K (row-major, b64 stores) and V (transposed, pair-packed b32)
    {
      const float* Kb = Kg + base + (size_t)kb * KTILE * D_DIM;
      const float* Vb = Vg + base + (size_t)kb * KTILE * D_DIM;
      for (int i = tid; i < KTILE * (D_DIM / 4); i += 256) {   // 16 iters/thread
        int row = i >> 5;
        int c4  = (i & 31) << 2;
        float4 fk = *(const float4*)(Kb + (size_t)row * D_DIM + c4);
        uint2 kk;
        kk.x = pkbf(fk.x, fk.y);
        kk.y = pkbf(fk.z, fk.w);
        *(uint2*)&sK[row * PADW + c4] = kk;
      }
      for (int i = tid; i < (KTILE / 2) * (D_DIM / 4); i += 256) { // 8 iters/thread
        int rp = i >> 5;                 // key row pair 0..63
        int c4 = (i & 31) << 2;          // d column 0..124
        float4 a = *(const float4*)(Vb + (size_t)(2 * rp) * D_DIM + c4);
        float4 b = *(const float4*)(Vb + (size_t)(2 * rp + 1) * D_DIM + c4);
        *(unsigned*)&sVt[(c4 + 0) * PADW + 2 * rp] = pkbf(a.x, b.x);
        *(unsigned*)&sVt[(c4 + 1) * PADW + 2 * rp] = pkbf(a.y, b.y);
        *(unsigned*)&sVt[(c4 + 2) * PADW + 2 * rp] = pkbf(a.z, b.z);
        *(unsigned*)&sVt[(c4 + 3) * PADW + 2 * rp] = pkbf(a.w, b.w);
      }
      if (kb < qtile) {  // prefetch next block into L2 (global_prefetch_b8)
        const float* Kn = Kb + (size_t)KTILE * D_DIM;
        const float* Vn = Vb + (size_t)KTILE * D_DIM;
        __builtin_prefetch(Kn + tid * 64, 0, 0);
        __builtin_prefetch(Vn + tid * 64, 0, 0);
      }
    }
    __syncthreads();

    const bool maskblk = (kb == qtile);
    const int  nch     = maskblk ? (((w * 16 + 15) >> 6) + 1) : 2;  // 64-key chunks

    for (int c = 0; c < nch; ++c) {
      const int kc = c * 64;
      // ---- S^T tiles: C[m=key][n=q], four 16-key tiles, K-dim = D = 128
      v8f ct[4];
#pragma unroll
      for (int t = 0; t < 4; ++t) ct[t] = zero;
#pragma unroll
      for (int j = 0; j < 4; ++j) {
#pragma unroll
        for (int t = 0; t < 4; ++t) {
          v16bf ak = ldsFragA(sK, kc + 16 * t + lr, 32 * j, h);
          ct[t] = __builtin_amdgcn_wmma_f32_16x16x32_bf16(false, ak, false, qf[j].v,
                                                          (short)0, ct[t], false, false);
        }
      }
      float s[4][8];
#pragma unroll
      for (int t = 0; t < 4; ++t)
#pragma unroll
        for (int r = 0; r < 8; ++r) s[t][r] = ct[t][r];
      if (maskblk) {
        const int key0 = kb * KTILE + kc + sb;   // abs key of (tile0, r=0), this half
#pragma unroll
        for (int t = 0; t < 4; ++t)
#pragma unroll
          for (int r = 0; r < 8; ++r)
            if (key0 + 16 * t + r > q_abs) s[t][r] = NEGBIG;
      }
      // ---- online softmax (per-lane state; halves combined via xor-16 shuffle)
      float mloc = s[0][0];
#pragma unroll
      for (int t = 0; t < 4; ++t)
#pragma unroll
        for (int r = 0; r < 8; ++r) mloc = fmaxf(mloc, s[t][r]);
      mloc = fmaxf(mloc, __shfl_xor(mloc, 16, 32));
      const float mnew  = fmaxf(m, mloc);
      const float alpha = exp2f((m - mnew) * LOG2E);
      float p[4][8], csum = 0.0f;
#pragma unroll
      for (int t = 0; t < 4; ++t)
#pragma unroll
        for (int r = 0; r < 8; ++r) {
          p[t][r] = exp2f((s[t][r] - mnew) * LOG2E);
          csum += p[t][r];
        }
      csum += __shfl_xor(csum, 16, 32);
      lsum = lsum * alpha + csum;
      m = mnew;

      // ---- pack P into two A-fragments (layouts line up, no cross-lane moves)
      FragBF ap0, ap1;
#pragma unroll
      for (int j = 0; j < 4; ++j) {
        ap0.u[j]     = pkbf(p[0][2 * j], p[0][2 * j + 1]);
        ap0.u[4 + j] = pkbf(p[1][2 * j], p[1][2 * j + 1]);
        ap1.u[j]     = pkbf(p[2][2 * j], p[2][2 * j + 1]);
        ap1.u[4 + j] = pkbf(p[3][2 * j], p[3][2 * j + 1]);
      }
      // ---- broadcast per-q alpha into O layout, rescale once per 64 keys
      float aR[8];
#pragma unroll
      for (int r = 0; r < 8; ++r) aR[r] = __shfl(alpha, sb + r, 32);
#pragma unroll
      for (int dt = 0; dt < 8; ++dt) {
        v8f acc = o[dt];
#pragma unroll
        for (int r = 0; r < 8; ++r) acc[r] *= aR[r];
        v16bf bv0 = ldsFragB(sVt, dt * 16 + lr, kc + 8 * h);
        acc = __builtin_amdgcn_wmma_f32_16x16x32_bf16(false, ap0.v, false, bv0,
                                                      (short)0, acc, false, false);
        v16bf bv1 = ldsFragB(sVt, dt * 16 + lr, kc + 32 + 8 * h);
        o[dt] = __builtin_amdgcn_wmma_f32_16x16x32_bf16(false, ap1.v, false, bv1,
                                                        (short)0, acc, false, false);
      }
    }
  }

  // ---- epilogue: normalize by 1/l and store
  float linv = 1.0f / lsum;
  float lR[8];
#pragma unroll
  for (int r = 0; r < 8; ++r) lR[r] = __shfl(linv, sb + r, 32);
  float* outp = Og + base + (size_t)(qtile * QTILE + w * 16) * D_DIM;
#pragma unroll
  for (int dt = 0; dt < 8; ++dt)
#pragma unroll
    for (int r = 0; r < 8; ++r)
      outp[(size_t)(r + sb) * D_DIM + dt * 16 + lr] = o[dt][r] * lR[r];
}

extern "C" void kernel_launch(void* const* d_in, const int* in_sizes, int n_in,
                              void* d_out, int out_size, void* d_ws, size_t ws_size,
                              hipStream_t stream) {
  (void)in_sizes; (void)n_in; (void)out_size; (void)d_ws; (void)ws_size;
  const float* Q = (const float*)d_in[0];
  const float* K = (const float*)d_in[1];
  const float* V = (const float*)d_in[2];
  // d_in[3] = attn_mask (bool causal) — implied by kernel's causal logic
  float* O = (float*)d_out;
  dim3 grid(S_LEN / QTILE, 2 * 16);   // (q tiles, B*H)
  fa_fwd_causal<<<grid, 256, 0, stream>>>(Q, K, V, O);
}
